// Encoder_57054345560074
// MI455X (gfx1250) — compile-verified, matrix operand order checked by reading, not run on hardware
//
#include <hip/hip_runtime.h>
#include <hip/hip_bf16.h>

typedef _Float16 f16;
typedef __attribute__((ext_vector_type(16))) _Float16 v16h;
typedef __attribute__((ext_vector_type(8)))  _Float16 h8;
typedef __attribute__((ext_vector_type(8)))  float    v8f;

#define B_    64
#define T_    256
#define H_    768
#define G4H   3072
#define MEL_  80
#define KP    96     // MEL padded to multiple of 32
#define E_    256
#define S_    3

// ---------------------------------------------------------------------------
// 16x32 f16 operand tile load (A or pre-transposed B), per CDNA5 wave32 layout:
// lanes 0-15 : row = lane,    K offsets {0..7, 16..23}
// lanes 16-31: row = lane-16, K offsets {8..15, 24..31}
// base points at element (row0, k0) of a row-major [rows x ld] f16 matrix.
// ---------------------------------------------------------------------------
__device__ __forceinline__ v16h load_tile16(const f16* __restrict__ base,
                                            int ld, int lane) {
    int half = lane >> 4;          // 0 or 1
    int r    = lane & 15;
    const f16* p = base + r * ld + half * 8;
    h8 lo = *(const h8*)(p);       // K + {0..7} or {8..15}
    h8 hi = *(const h8*)(p + 16);  // K + {16..23} or {24..31}
    return __builtin_shufflevector(lo, hi, 0,1,2,3,4,5,6,7,8,9,10,11,12,13,14,15);
}

__device__ __forceinline__ float sigmoidf_(float x) {
    return 1.0f / (1.0f + __expf(-x));
}

#define WMMA_F16(a, b, c) \
    __builtin_amdgcn_wmma_f32_16x16x32_f16(false, (a), false, (b), (short)0, (c), false, false)

// ---------------------------------------------------------------------------
// Generic wave-level WMMA GEMM:  Out[M,N] = A[M,K] * Bw[N,K]^T + bias[N]
// Each wave computes a 16M x 64N tile (4 accumulators, A reused 4x).
// ---------------------------------------------------------------------------
template <bool OUT_F32>
__global__ __launch_bounds__(256, 1)
void gemm_wave(const f16* __restrict__ A, int lda,
               const f16* __restrict__ Bw, int ldb,
               const float* __restrict__ bias,
               void* __restrict__ Out, int ldo,
               int Mtiles, int Ngroups, int Kchunks) {
    int lane = threadIdx.x & 31;
    int w    = (blockIdx.x * blockDim.x + threadIdx.x) >> 5;
    if (w >= Mtiles * Ngroups) return;      // wave-uniform
    int mt = w / Ngroups;
    int ng = w % Ngroups;

    v8f acc0 = {}, acc1 = {}, acc2 = {}, acc3 = {};
    const f16* Abase = A + mt * 16 * lda;
    const f16* Bbase = Bw + ng * 64 * ldb;

    for (int kc = 0; kc < Kchunks; ++kc) {
        int k0 = kc * 32;
        __builtin_prefetch(Bbase + (k0 + 32), 0, 0);
        v16h a  = load_tile16(Abase + k0, lda, lane);
        v16h b0 = load_tile16(Bbase + 0  * ldb + k0, ldb, lane);
        v16h b1 = load_tile16(Bbase + 16 * ldb + k0, ldb, lane);
        v16h b2 = load_tile16(Bbase + 32 * ldb + k0, ldb, lane);
        v16h b3 = load_tile16(Bbase + 48 * ldb + k0, ldb, lane);
        acc0 = WMMA_F16(a, b0, acc0);
        acc1 = WMMA_F16(a, b1, acc1);
        acc2 = WMMA_F16(a, b2, acc2);
        acc3 = WMMA_F16(a, b3, acc3);
    }

    // C/D layout: lanes 0-15 -> M = r, lanes 16-31 -> M = 8 + r; N = lane&15
    int mrow0 = mt * 16 + ((lane >> 4) << 3);
    int ncol  = lane & 15;

    auto store_tile = [&](v8f acc, int ntile) {
        int n = ng * 64 + ntile * 16 + ncol;
        float bv = bias ? bias[n] : 0.0f;
#pragma unroll
        for (int r = 0; r < 8; ++r) {
            int m = mrow0 + r;
            float v = acc[r] + bv;
            if (OUT_F32) ((float*)Out)[(size_t)m * ldo + n] = v;
            else         ((f16*)Out)[(size_t)m * ldo + n] = (f16)v;
        }
    };
    store_tile(acc0, 0);
    store_tile(acc1, 1);
    store_tile(acc2, 2);
    store_tile(acc3, 3);
}

// ---------------------------------------------------------------------------
// One LSTM timestep. Grid = 96 single-wave blocks:
//   col group j = blockIdx.x % 48  -> hidden columns hc = 16j .. 16j+15
//   batch half  = blockIdx.x / 48  -> batch rows 32*half .. 32*half+31
// Each wave holds 2 M-tiles x 4 gates = 8 f32 accumulators (64 VGPRs),
// named variables to avoid spills. i/f/g/o live at columns n, H+n, 2H+n, 3H+n
// so the LSTM elementwise math is lane-local. h ping-pongs across launches.
// gates = hin[64,768] @ Whh[3072,768]^T + pre[:,t,:]
// ---------------------------------------------------------------------------
__global__ __launch_bounds__(32, 1)
void lstm_step(const f16* __restrict__ hin,     // [64,768]
               f16* __restrict__ hout,          // [64,768]
               const f16* __restrict__ pre_t,   // pre + t*4H ; batch stride T*4H
               const f16* __restrict__ Whh,     // [3072,768]
               float* __restrict__ cbuf,        // [64,768]
               f16* __restrict__ y_t) {         // y + t*H ; batch stride T*H
    int lane  = threadIdx.x & 31;
    int j     = blockIdx.x % (H_ / 16);
    int mhalf = blockIdx.x / (H_ / 16);         // 0 or 1
    int hc    = j * 16;
    int mbase = mhalf * 32;                     // first batch row of this wave

    v8f ci0 = {}, cf0 = {}, cg0 = {}, co0 = {};   // m-tile 0 (rows mbase..mbase+15)
    v8f ci1 = {}, cf1 = {}, cg1 = {}, co1 = {};   // m-tile 1 (rows mbase+16..+31)

    const f16* Wi = Whh + (size_t)(0 * H_ + hc) * H_;
    const f16* Wf = Whh + (size_t)(1 * H_ + hc) * H_;
    const f16* Wg = Whh + (size_t)(2 * H_ + hc) * H_;
    const f16* Wo = Whh + (size_t)(3 * H_ + hc) * H_;

#pragma unroll 1
    for (int kc = 0; kc < H_ / 32; ++kc) {
        int k0 = kc * 32;
        __builtin_prefetch(Wi + k0 + 32, 0, 0);
        v16h a0 = load_tile16(hin + (size_t)(mbase +  0) * H_ + k0, H_, lane);
        v16h a1 = load_tile16(hin + (size_t)(mbase + 16) * H_ + k0, H_, lane);

        v16h b;
        b = load_tile16(Wi + k0, H_, lane);
        ci0 = WMMA_F16(a0, b, ci0);  ci1 = WMMA_F16(a1, b, ci1);
        b = load_tile16(Wf + k0, H_, lane);
        cf0 = WMMA_F16(a0, b, cf0);  cf1 = WMMA_F16(a1, b, cf1);
        b = load_tile16(Wg + k0, H_, lane);
        cg0 = WMMA_F16(a0, b, cg0);  cg1 = WMMA_F16(a1, b, cg1);
        b = load_tile16(Wo + k0, H_, lane);
        co0 = WMMA_F16(a0, b, co0);  co1 = WMMA_F16(a1, b, co1);
    }

    int n     = hc + (lane & 15);
    int mrofs = (lane >> 4) << 3;

    auto epilogue = [&](v8f ai, v8f af, v8f ag, v8f ao, int mtrow0) {
#pragma unroll
        for (int r = 0; r < 8; ++r) {
            int m = mtrow0 + mrofs + r;
            size_t pb = (size_t)m * (T_ * G4H);
            float iv = ai[r] + (float)pre_t[pb + 0 * H_ + n];
            float fv = af[r] + (float)pre_t[pb + 1 * H_ + n];
            float gv = ag[r] + (float)pre_t[pb + 2 * H_ + n];
            float ov = ao[r] + (float)pre_t[pb + 3 * H_ + n];
            float c_old = cbuf[(size_t)m * H_ + n];
            float c_new = sigmoidf_(fv) * c_old + sigmoidf_(iv) * tanhf(gv);
            float h_new = sigmoidf_(ov) * tanhf(c_new);
            cbuf[(size_t)m * H_ + n] = c_new;
            hout[(size_t)m * H_ + n] = (f16)h_new;
            y_t[(size_t)m * (T_ * H_) + n] = (f16)h_new;
        }
    };
    epilogue(ci0, cf0, cg0, co0, mbase);
    epilogue(ci1, cf1, cg1, co1, mbase + 16);
}

// ---------------------------------------------------------------------------
// Prep / elementwise kernels
// ---------------------------------------------------------------------------
__global__ void cvt_f32_to_f16(const float* __restrict__ in, f16* __restrict__ out, int n) {
    int i = blockIdx.x * blockDim.x + threadIdx.x;
    if (i < n) out[i] = (f16)in[i];
}

__global__ void pad_prenet_w(const float* __restrict__ in, f16* __restrict__ out) {
    int i = blockIdx.x * blockDim.x + threadIdx.x;     // over H_*KP
    if (i >= H_ * KP) return;
    int h = i / KP, m = i % KP;
    out[i] = (m < MEL_) ? (f16)in[h * MEL_ + m] : (f16)0.0f;
}

__global__ void bias_sum(const float* __restrict__ a, const float* __restrict__ b,
                         float* __restrict__ out, int n) {
    int i = blockIdx.x * blockDim.x + threadIdx.x;
    if (i < n) out[i] = a[i] + b[i];
}

// mels [B, MEL, T] fp32 -> xm [(B*T), KP] f16 (K-padded with zeros)
__global__ void mels_transpose(const float* __restrict__ mels, f16* __restrict__ xm) {
    int i = blockIdx.x * blockDim.x + threadIdx.x;     // over B*T*KP
    if (i >= B_ * T_ * KP) return;
    int m  = i % KP;
    int bt = i / KP;
    int b  = bt / T_, t = bt % T_;
    xm[i] = (m < MEL_) ? (f16)mels[((size_t)b * MEL_ + m) * T_ + t] : (f16)0.0f;
}

__global__ void zero_u32(unsigned* __restrict__ p, int nwords) {
    int i = blockIdx.x * blockDim.x + threadIdx.x;
    if (i < nwords) p[i] = 0u;
}

// y += x (f16, 8 elems per thread)
__global__ void residual_add(f16* __restrict__ y, const f16* __restrict__ x, int n) {
    int i = (blockIdx.x * blockDim.x + threadIdx.x) * 8;
    if (i + 8 > n) return;
#pragma unroll
    for (int k = 0; k < 8; ++k)
        y[i + k] = (f16)((float)y[i + k] + (float)x[i + k]);
}

// ---------------------------------------------------------------------------
// Host launcher
// ---------------------------------------------------------------------------
extern "C" void kernel_launch(void* const* d_in, const int* in_sizes, int n_in,
                              void* d_out, int out_size, void* d_ws, size_t ws_size,
                              hipStream_t stream) {
    const float* mels     = (const float*)d_in[0];
    const float* prenet_W = (const float*)d_in[1];
    const float* prenet_b = (const float*)d_in[2];
    const float* W_ih     = (const float*)d_in[3];
    const float* W_hh     = (const float*)d_in[4];
    const float* b_ih     = (const float*)d_in[5];
    const float* b_hh     = (const float*)d_in[6];
    const float* out_W    = (const float*)d_in[7];
    const float* out_b    = (const float*)d_in[8];

    // ---- workspace carve-up (bytes) ----
    char* ws = (char*)d_ws;
    size_t off = 0;
    auto alloc = [&](size_t bytes) { void* p = ws + off; off += (bytes + 255) & ~(size_t)255; return p; };
    f16*   wPre = (f16*)  alloc((size_t)H_ * KP * 2);              // 768x96
    f16*   wIh  = (f16*)  alloc((size_t)S_ * G4H * H_ * 2);        // 3x3072x768
    f16*   wHh  = (f16*)  alloc((size_t)S_ * G4H * H_ * 2);
    f16*   wOut = (f16*)  alloc((size_t)E_ * H_ * 2);
    float* bsum = (float*)alloc((size_t)S_ * G4H * 4);
    f16*   xm   = (f16*)  alloc((size_t)B_ * T_ * KP * 2);
    f16*   xb   = (f16*)  alloc((size_t)B_ * T_ * H_ * 2);
    f16*   yb   = (f16*)  alloc((size_t)B_ * T_ * H_ * 2);
    f16*   pre  = (f16*)  alloc((size_t)B_ * T_ * G4H * 2);        // ~100 MB
    f16*   h0   = (f16*)  alloc((size_t)B_ * H_ * 2);
    f16*   h1   = (f16*)  alloc((size_t)B_ * H_ * 2);
    float* cbuf = (float*)alloc((size_t)B_ * H_ * 4);
    (void)ws_size;

    const int nWih = S_ * G4H * H_;      // 7,077,888
    const int nWout = E_ * H_;           // 196,608

    // ---- weight prep ----
    cvt_f32_to_f16<<<(nWih + 255) / 256, 256, 0, stream>>>(W_ih, wIh, nWih);
    cvt_f32_to_f16<<<(nWih + 255) / 256, 256, 0, stream>>>(W_hh, wHh, nWih);
    cvt_f32_to_f16<<<(nWout + 255) / 256, 256, 0, stream>>>(out_W, wOut, nWout);
    pad_prenet_w<<<(H_ * KP + 255) / 256, 256, 0, stream>>>(prenet_W, wPre);
    bias_sum<<<(S_ * G4H + 255) / 256, 256, 0, stream>>>(b_ih, b_hh, bsum, S_ * G4H);
    mels_transpose<<<(B_ * T_ * KP + 255) / 256, 256, 0, stream>>>(mels, xm);

    const int MT = (B_ * T_) / 16;       // 1024 M-tiles

    // ---- prenet: xb[BT,768] = xm @ wPre^T + prenet_b ----
    {
        int waves = MT * (H_ / 64);      // 1024 * 12
        gemm_wave<false><<<(waves + 7) / 8, 256, 0, stream>>>(
            xm, KP, wPre, KP, prenet_b, xb, H_, MT, H_ / 64, KP / 32);
    }

    f16* XB = xb;
    f16* YB = yb;
    for (int s = 0; s < S_; ++s) {
        // pre[BT,3072] = XB @ W_ih[s]^T + (b_ih+b_hh)
        int waves = MT * (G4H / 64);     // 1024 * 48
        gemm_wave<false><<<(waves + 7) / 8, 256, 0, stream>>>(
            XB, H_, wIh + (size_t)s * G4H * H_, H_, bsum + s * G4H,
            pre, G4H, MT, G4H / 64, H_ / 32);

        // zero h0 and c
        zero_u32<<<((B_ * H_ / 2) + 255) / 256, 256, 0, stream>>>((unsigned*)h0, B_ * H_ / 2);
        zero_u32<<<((B_ * H_) + 255) / 256, 256, 0, stream>>>((unsigned*)cbuf, B_ * H_);

        // sequential recurrence: launch-per-timestep is the grid-wide barrier
        for (int t = 0; t < T_; ++t) {
            const f16* hin = (t & 1) ? h1 : h0;
            f16* hout      = (t & 1) ? h0 : h1;
            lstm_step<<<2 * (H_ / 16), 32, 0, stream>>>(
                hin, hout,
                pre + (size_t)t * G4H,
                wHh + (size_t)s * G4H * H_,
                cbuf,
                YB + (size_t)t * H_);
        }
        // T_=256 is even -> final h landed in h0

        if (s < S_ - 1) {
            int n = B_ * T_ * H_;
            residual_add<<<(n / 8 + 255) / 256, 256, 0, stream>>>(YB, XB, n);
            f16* tmp = XB; XB = YB; YB = tmp;   // next layer input = y + x
        }
    }

    // ---- output projection: d_out[64,256] = h0 @ out_W^T + out_b (fp32) ----
    {
        int waves = (B_ / 16) * (E_ / 64);   // 4 * 4
        gemm_wave<true><<<(waves + 7) / 8, 256, 0, stream>>>(
            h0, H_, wOut, H_, out_b, d_out, E_, B_ / 16, E_ / 64, H_ / 32);
    }
    (void)in_sizes; (void)n_in; (void)out_size;
}
